// ScaledDotProductAttention_26216480374997
// MI455X (gfx1250) — matrix operand order repaired and checked by reading, not compiled, not run on hardware
//
#include <hip/hip_runtime.h>
#include <hip/hip_bf16.h>

#define B_ 4
#define H_ 16
#define S_ 2048
#define DK_ 64
#define NELEM (B_*H_*S_*DK_)   // 8388608 per tensor

typedef __attribute__((ext_vector_type(16))) __bf16 v16bf;
typedef __attribute__((ext_vector_type(4)))  __bf16 v4bf;
typedef __attribute__((ext_vector_type(8)))  float  v8f;
typedef __attribute__((ext_vector_type(4)))  unsigned int v4u;

union ABFrag { v16bf bf; v4u u[2]; };

static __device__ __forceinline__ v8f wmma_bf16(const ABFrag& a, const ABFrag& b, v8f c) {
  // 8 args: (neg_a, A, neg_b, B, c_mod, C, reuse_a, reuse_b)
  return __builtin_amdgcn_wmma_f32_16x16x32_bf16(false, a.bf, false, b.bf, (short)0, c, false, false);
}

// ---- DPP 16-lane butterfly reductions (VALU only; no DS pipe, no dscnt waits)
// ctrl: 0xB1 quad_perm(1,0,3,2)=xor1, 0x4E quad_perm(2,3,0,1)=xor2,
//       0x141 row_half_mirror (xor4 once quads uniform), 0x140 row_mirror (xor8)
template <int CTRL>
static __device__ __forceinline__ float dpp_bcast(float x) {
  int r = __builtin_amdgcn_update_dpp(0, __builtin_bit_cast(int, x), CTRL, 0xF, 0xF, true);
  return __builtin_bit_cast(float, r);
}
static __device__ __forceinline__ float rowmax16(float t) {
  t = fmaxf(t, dpp_bcast<0xB1>(t));
  t = fmaxf(t, dpp_bcast<0x4E>(t));
  t = fmaxf(t, dpp_bcast<0x141>(t));
  t = fmaxf(t, dpp_bcast<0x140>(t));
  return t;
}
static __device__ __forceinline__ float rowsum16(float t) {
  t += dpp_bcast<0xB1>(t);
  t += dpp_bcast<0x4E>(t);
  t += dpp_bcast<0x141>(t);
  t += dpp_bcast<0x140>(t);
  return t;
}

// ---------------------------------------------------------------------------
// Pass 1: fp32 -> bf16 convert. Q is pre-scaled by (1/sqrt(DK)) * log2(e) so
// the softmax runs in the exp2 domain (v_exp_f32 with no extra multiply).
// ---------------------------------------------------------------------------
__global__ __launch_bounds__(256) void cvt_qk_kernel(const float* __restrict__ q,
                                                     const float* __restrict__ k,
                                                     __bf16* __restrict__ qb,
                                                     __bf16* __restrict__ kb) {
  const float scale = 0.125f * 1.44269504088896f;  // 1/sqrt(64) * log2(e)
  size_t i = (size_t)blockIdx.x * 256 + threadIdx.x;  // index of 4-element group
  float4 q4 = ((const float4*)q)[i];
  float4 k4 = ((const float4*)k)[i];
  v4bf qo, ko;
  qo.x = (__bf16)(q4.x * scale); qo.y = (__bf16)(q4.y * scale);
  qo.z = (__bf16)(q4.z * scale); qo.w = (__bf16)(q4.w * scale);
  ko.x = (__bf16)k4.x; ko.y = (__bf16)k4.y; ko.z = (__bf16)k4.z; ko.w = (__bf16)k4.w;
  *(v4bf*)(qb + 4*i) = qo;
  *(v4bf*)(kb + 4*i) = ko;
}

// ---------------------------------------------------------------------------
// Pass 2: V [b,h,s,d] fp32 -> Vt [b,h,d,s] bf16, LDS-tiled transpose.
// ---------------------------------------------------------------------------
__global__ __launch_bounds__(256) void transpose_v_kernel(const float* __restrict__ v,
                                                          __bf16* __restrict__ vt) {
  __shared__ __bf16 tile[64 * 66];
  const int bh = blockIdx.x >> 5;        // 32 s-tiles per bh
  const int st = blockIdx.x & 31;
  const float* src = v  + (size_t)bh * S_ * DK_ + (size_t)st * 64 * DK_;
  __bf16*      dst = vt + (size_t)bh * S_ * DK_ + (size_t)st * 64;   // [d][s]
  #pragma unroll
  for (int it = 0; it < 16; ++it) {
    int idx = threadIdx.x + it * 256;
    int s = idx >> 6, d = idx & 63;
    tile[s * 66 + d] = (__bf16)src[(size_t)s * DK_ + d];
  }
  __syncthreads();
  #pragma unroll
  for (int it = 0; it < 16; ++it) {
    int idx = threadIdx.x + it * 256;
    int d = idx >> 6, s = idx & 63;
    dst[(size_t)d * S_ + s] = tile[s * 66 + d];
  }
}

// ---------------------------------------------------------------------------
// Pass 3: flash attention. One wave owns a 16-query tile, streams keys in
// blocks of 64 (4 score sub-tiles). 16 bf16 WMMAs per iteration.
// ---------------------------------------------------------------------------
__global__ __launch_bounds__(256) void fa_kernel(const __bf16* __restrict__ qb,
                                                 const __bf16* __restrict__ kb,
                                                 const __bf16* __restrict__ vt,
                                                 const int*    __restrict__ mask,
                                                 float*        __restrict__ out) {
  __shared__ __bf16 pbuf[8][16 * 64];    // per-wave 16x64 P staging tile (2 KB)
  const int wave = threadIdx.x >> 5;
  const int lane = threadIdx.x & 31;
  const int half = lane >> 4;
  const int l16  = lane & 15;
  const int bh   = blockIdx.x >> 4;      // 16 q-blocks (of 128) per bh
  const int qblk = blockIdx.x & 15;
  const int b    = bh / H_;
  const int q0   = qblk * 128 + wave * 16;

  const __bf16* Q  = qb + (size_t)bh * S_ * DK_;
  const __bf16* K  = kb + (size_t)bh * S_ * DK_;
  const __bf16* VT = vt + (size_t)bh * S_ * DK_;   // [d][s]
  const int*   msk = mask + (size_t)b * S_;
  __bf16* plds = pbuf[wave];

  // Q tile as two A fragments (K-dim 0..31 / 32..63).
  ABFrag a0, a1;
  {
    const __bf16* qrow = Q + (size_t)(q0 + l16) * DK_ + half * 8;
    a0.u[0] = *(const v4u*)(qrow);
    a0.u[1] = *(const v4u*)(qrow + 16);
    a1.u[0] = *(const v4u*)(qrow + 32);
    a1.u[1] = *(const v4u*)(qrow + 48);
  }

  float m[8], l[8];
  #pragma unroll
  for (int r = 0; r < 8; ++r) { m[r] = -1e30f; l[r] = 0.0f; }
  v8f o0 = {}, o1 = {}, o2 = {}, o3 = {};

  for (int kt = 0; kt < S_; kt += 64) {
    // ---- scores for 4 key sub-tiles (16 keys each)
    v8f s0 = {}, s1 = {}, s2 = {}, s3 = {};
    {
      const __bf16* krow = K + (size_t)(kt + l16) * DK_ + half * 8;
      ABFrag ba, bb;
      ba.u[0] = *(const v4u*)(krow);      ba.u[1] = *(const v4u*)(krow + 16);
      bb.u[0] = *(const v4u*)(krow + 32); bb.u[1] = *(const v4u*)(krow + 48);
      s0 = wmma_bf16(a0, ba, s0); s0 = wmma_bf16(a1, bb, s0);
      krow += 16 * DK_;
      ba.u[0] = *(const v4u*)(krow);      ba.u[1] = *(const v4u*)(krow + 16);
      bb.u[0] = *(const v4u*)(krow + 32); bb.u[1] = *(const v4u*)(krow + 48);
      s1 = wmma_bf16(a0, ba, s1); s1 = wmma_bf16(a1, bb, s1);
      krow += 16 * DK_;
      ba.u[0] = *(const v4u*)(krow);      ba.u[1] = *(const v4u*)(krow + 16);
      bb.u[0] = *(const v4u*)(krow + 32); bb.u[1] = *(const v4u*)(krow + 48);
      s2 = wmma_bf16(a0, ba, s2); s2 = wmma_bf16(a1, bb, s2);
      krow += 16 * DK_;
      ba.u[0] = *(const v4u*)(krow);      ba.u[1] = *(const v4u*)(krow + 16);
      bb.u[0] = *(const v4u*)(krow + 32); bb.u[1] = *(const v4u*)(krow + 48);
      s3 = wmma_bf16(a0, ba, s3); s3 = wmma_bf16(a1, bb, s3);
    }

    // ---- mask bias (column per lane: n = l16 + 16*tile); SLP-packs to v_pk_add_f32
    const float bias0 = msk[kt + l16]      ? 0.0f : -1e30f;
    const float bias1 = msk[kt + 16 + l16] ? 0.0f : -1e30f;
    const float bias2 = msk[kt + 32 + l16] ? 0.0f : -1e30f;
    const float bias3 = msk[kt + 48 + l16] ? 0.0f : -1e30f;
    #pragma unroll
    for (int r = 0; r < 8; ++r) {
      s0[r] += bias0; s1[r] += bias1; s2[r] += bias2; s3[r] += bias3;
    }

    // ---- phase 1: per-row tile maxes + DPP butterflies (row-uniform result)
    float t[8];
    #pragma unroll
    for (int r = 0; r < 8; ++r)
      t[r] = rowmax16(fmaxf(fmaxf(s0[r], s1[r]), fmaxf(s2[r], s3[r])));

    // ---- phase 2: wave-uniform skip-rescale. alpha==1 exactly when no row's
    // max grew, so the whole rescale block can be skipped (bit-exact).
    float need = t[0] - m[0];
    #pragma unroll
    for (int r = 1; r < 8; ++r) need = fmaxf(need, t[r] - m[r]);
    if (__any(need > 0.0f)) {
      float alpha[8];
      #pragma unroll
      for (int r = 0; r < 8; ++r) {
        const float mn = fmaxf(m[r], t[r]);
        alpha[r] = exp2f(m[r] - mn);
        m[r] = mn;
      }
      #pragma unroll
      for (int r = 0; r < 8; ++r) l[r] *= alpha[r];
      #pragma unroll
      for (int r = 0; r < 8; ++r) o0[r] *= alpha[r];
      #pragma unroll
      for (int r = 0; r < 8; ++r) o1[r] *= alpha[r];
      #pragma unroll
      for (int r = 0; r < 8; ++r) o2[r] *= alpha[r];
      #pragma unroll
      for (int r = 0; r < 8; ++r) o3[r] *= alpha[r];
    }

    // ---- phase 3: exponentials + lazy per-lane row sums (TRANS co-executes)
    #pragma unroll
    for (int r = 0; r < 8; ++r) {
      const float p0 = exp2f(s0[r] - m[r]);
      const float p1 = exp2f(s1[r] - m[r]);
      const float p2 = exp2f(s2[r] - m[r]);
      const float p3 = exp2f(s3[r] - m[r]);
      s0[r] = p0; s1[r] = p1; s2[r] = p2; s3[r] = p3;
      l[r] += (p0 + p1) + (p2 + p3);
    }

    // ---- stage P (16x64 bf16, row-major in K) through per-wave LDS
    #pragma unroll
    for (int r = 0; r < 8; ++r) {
      const int row = r + 8 * half;
      __bf16* pr = plds + row * 64 + l16;
      pr[0]  = (__bf16)s0[r];
      pr[16] = (__bf16)s1[r];
      pr[32] = (__bf16)s2[r];
      pr[48] = (__bf16)s3[r];
    }
    asm volatile("s_wait_dscnt 0" ::: "memory");
    ABFrag pa0, pa1;
    {
      const __bf16* prow = plds + l16 * 64 + half * 8;
      pa0.u[0] = *(const v4u*)(prow);
      pa0.u[1] = *(const v4u*)(prow + 16);
      pa1.u[0] = *(const v4u*)(prow + 32);
      pa1.u[1] = *(const v4u*)(prow + 48);
    }

    // ---- O += P @ V  (two 32-key chunks per head-dim tile)
    #pragma unroll
    for (int nt = 0; nt < 4; ++nt) {
      ABFrag vb0, vb1;
      const __bf16* vrow = VT + (size_t)(nt * 16 + l16) * S_ + kt + half * 8;
      vb0.u[0] = *(const v4u*)(vrow);      vb0.u[1] = *(const v4u*)(vrow + 16);
      vb1.u[0] = *(const v4u*)(vrow + 32); vb1.u[1] = *(const v4u*)(vrow + 48);
      if      (nt == 0) { o0 = wmma_bf16(pa0, vb0, o0); o0 = wmma_bf16(pa1, vb1, o0); }
      else if (nt == 1) { o1 = wmma_bf16(pa0, vb0, o1); o1 = wmma_bf16(pa1, vb1, o1); }
      else if (nt == 2) { o2 = wmma_bf16(pa0, vb0, o2); o2 = wmma_bf16(pa1, vb1, o2); }
      else              { o3 = wmma_bf16(pa0, vb0, o3); o3 = wmma_bf16(pa1, vb1, o3); }
    }
  }

  // ---- finalize: one cross-lane sum reduce, divide, write fp32 output
  float* orow = out + ((size_t)bh * S_ + q0) * DK_;
  #pragma unroll
  for (int r = 0; r < 8; ++r) {
    const float inv = 1.0f / rowsum16(l[r]);
    float* po = orow + (size_t)(r + 8 * half) * DK_;
    po[l16]      = o0[r] * inv;
    po[16 + l16] = o1[r] * inv;
    po[32 + l16] = o2[r] * inv;
    po[48 + l16] = o3[r] * inv;
  }
}

// ---------------------------------------------------------------------------
extern "C" void kernel_launch(void* const* d_in, const int* in_sizes, int n_in,
                              void* d_out, int out_size, void* d_ws, size_t ws_size,
                              hipStream_t stream) {
  const float* q   = (const float*)d_in[0];
  const float* k   = (const float*)d_in[1];
  const float* v   = (const float*)d_in[2];
  const int* mask  = (const int*)d_in[3];
  float* out       = (float*)d_out;

  __bf16* qb = (__bf16*)d_ws;
  __bf16* kb = qb + (size_t)NELEM;
  __bf16* vt = kb + (size_t)NELEM;

  cvt_qk_kernel<<<NELEM / 4 / 256, 256, 0, stream>>>(q, k, qb, kb);
  transpose_v_kernel<<<B_ * H_ * (S_ / 64), 256, 0, stream>>>(v, vt);
  fa_kernel<<<B_ * H_ * (S_ / 128), 256, 0, stream>>>(qb, kb, vt, mask, out);
}
